// GAT_9294309229068
// MI455X (gfx1250) — matrix operand order repaired and checked by reading, not compiled, not run on hardware
//
#include <hip/hip_runtime.h>
#include <hip/hip_bf16.h>

// ---------------------------------------------------------------------------
// Types for CDNA5 WMMA (wave32): v_wmma_f32_16x16x32_bf16
// ---------------------------------------------------------------------------
typedef __attribute__((ext_vector_type(16))) __bf16 v16bf;
typedef __attribute__((ext_vector_type(8)))  __bf16 v8bf;
typedef __attribute__((ext_vector_type(8)))  float  v8f;

// ---------------------------------------------------------------------------
// Problem constants (from the reference)
// ---------------------------------------------------------------------------
#define NN    50000
#define NE    800000
#define FIN   256
#define NH    3
#define ND    64
#define NHD   192   // NH*ND
#define NOUT  64

// ---------------------------------------------------------------------------
// Order-preserving float <-> uint map for atomicMax-based segment max
// ---------------------------------------------------------------------------
__device__ __forceinline__ unsigned mapF(float x) {
    unsigned b = __float_as_uint(x);
    return (b & 0x80000000u) ? ~b : (b | 0x80000000u);
}
__device__ __forceinline__ float unmapF(unsigned u) {
    unsigned b = (u & 0x80000000u) ? (u & 0x7FFFFFFFu) : ~u;
    return __uint_as_float(b);
}

__device__ __forceinline__ v16bf pack16(v8bf lo, v8bf hi) {
    v16bf r;
#pragma unroll
    for (int i = 0; i < 8; ++i) { r[i] = lo[i]; r[i + 8] = hi[i]; }
    return r;
}

// ---------------------------------------------------------------------------
// bf16 WMMA GEMM:  C[M,N] = A[M,K](bf16) * Bt[N,K]^T(bf16) (+bias) (+relu)
// One wave per 16x64 output block (4 N-tiles share one A fragment -> 4 WMMAs
// per 10 b128 loads). Requires M%16==0, N%64==0, K%32==0.
// EXEC stays all-ones per wave (block guard is wave-uniform).
// ---------------------------------------------------------------------------
__global__ void gemm_bf16_wmma(const __bf16* __restrict__ A,
                               const __bf16* __restrict__ Bt,
                               const float*  __restrict__ bias,
                               float*        __restrict__ C,
                               int M, int N, int K, int act, int totalBlocks) {
    const int wave = blockIdx.x * (blockDim.x >> 5) + (threadIdx.x >> 5);
    if (wave >= totalBlocks) return;                // wave-uniform predicate
    const int ngroups = N >> 6;                     // 64-wide N blocks
    const int mt = wave / ngroups;
    const int g  = wave - mt * ngroups;

    const int lane = threadIdx.x & 31;
    const int hl   = lane >> 4;                     // half-wave select
    const int r    = lane & 15;

    const __bf16* aRow = A + (size_t)(mt * 16 + r) * K;
    const __bf16* bRow[4];
#pragma unroll
    for (int j = 0; j < 4; ++j)
        bRow[j] = Bt + (size_t)(g * 64 + j * 16 + r) * K;

    v8f acc0 = {}, acc1 = {}, acc2 = {}, acc3 = {};
    for (int k0 = 0; k0 < K; k0 += 32) {
        // Speculative prefetch of next A K-chunk (global_prefetch_b8)
        __builtin_prefetch(aRow + k0 + 32 + hl * 8, 0, 3);

        // A 16x32 bf16: lane<16 holds K 0..15; lane>=16 holds K 16..31
        v16bf a = pack16(*(const v8bf*)(aRow + k0 + hl * 16),
                         *(const v8bf*)(aRow + k0 + hl * 16 + 8));
        // B 32x16 bf16 (stored as Bt row-major): lane<16 -> K 0..15, lane>=16 -> K 16..31
        v16bf b0 = pack16(*(const v8bf*)(bRow[0] + k0 + hl * 16),
                          *(const v8bf*)(bRow[0] + k0 + hl * 16 + 8));
        v16bf b1 = pack16(*(const v8bf*)(bRow[1] + k0 + hl * 16),
                          *(const v8bf*)(bRow[1] + k0 + hl * 16 + 8));
        v16bf b2 = pack16(*(const v8bf*)(bRow[2] + k0 + hl * 16),
                          *(const v8bf*)(bRow[2] + k0 + hl * 16 + 8));
        v16bf b3 = pack16(*(const v8bf*)(bRow[3] + k0 + hl * 16),
                          *(const v8bf*)(bRow[3] + k0 + hl * 16 + 8));

        acc0 = __builtin_amdgcn_wmma_f32_16x16x32_bf16(false, a, false, b0, (short)0, acc0, false, false);
        acc1 = __builtin_amdgcn_wmma_f32_16x16x32_bf16(false, a, false, b1, (short)0, acc1, false, false);
        acc2 = __builtin_amdgcn_wmma_f32_16x16x32_bf16(false, a, false, b2, (short)0, acc2, false, false);
        acc3 = __builtin_amdgcn_wmma_f32_16x16x32_bf16(false, a, false, b3, (short)0, acc3, false, false);
    }

    // C/D layout: VGPR i -> row (i + hl*8), col = r (within each 16x16 tile)
    float* cBase = C + (size_t)(mt * 16 + hl * 8) * N + g * 64 + r;
#pragma unroll
    for (int j = 0; j < 4; ++j) {
        const v8f acc = (j == 0) ? acc0 : (j == 1) ? acc1 : (j == 2) ? acc2 : acc3;
        const int col = g * 64 + j * 16 + r;
        const float bv = bias ? bias[col] : 0.0f;
        float* cj = cBase + j * 16;
#pragma unroll
        for (int i = 0; i < 8; ++i) {
            float v = acc[i] + bv;
            if (act) v = fmaxf(v, 0.0f);
            cj[(size_t)i * N] = v;
        }
    }
}

// ---------------------------------------------------------------------------
// Elementwise helpers
// ---------------------------------------------------------------------------
__global__ void cvt_bf16(const float* __restrict__ in, __bf16* __restrict__ out, int n) {
    int i = blockIdx.x * blockDim.x + threadIdx.x;
    if (i < n) out[i] = (__bf16)in[i];
}

// W[K,N] (row-major fp32) -> Wt[N,K] (row-major bf16)
__global__ void tconv_bf16(const float* __restrict__ W, __bf16* __restrict__ Wt, int K, int N) {
    int i = blockIdx.x * blockDim.x + threadIdx.x;
    if (i >= K * N) return;
    int k = i / N, n = i - k * N;
    Wt[(size_t)n * K + k] = (__bf16)W[i];
}

__global__ void fill_zero(unsigned* __restrict__ p, int n) {
    int i = blockIdx.x * blockDim.x + threadIdx.x;
    if (i < n) p[i] = 0u;
}

// el/er: per (node, head) 64-length dot with attn vectors
__global__ void node_attn(const float* __restrict__ feat,
                          const float* __restrict__ al, const float* __restrict__ ar,
                          float* __restrict__ el, float* __restrict__ er, int nNodes) {
    int idx = blockIdx.x * blockDim.x + threadIdx.x;     // node*NH + h
    if (idx >= nNodes * NH) return;
    int nd = idx / NH, h = idx - nd * NH;
    const float* f = feat + (size_t)nd * NHD + h * ND;
    const float* a = al + h * ND;
    const float* b = ar + h * ND;
    float sl = 0.f, sr = 0.f;
#pragma unroll 8
    for (int d = 0; d < ND; ++d) { float v = f[d]; sl += v * a[d]; sr += v * b[d]; }
    el[idx] = sl; er[idx] = sr;
}

// e = leaky_relu(el[src]+er[dst]); segment max into mU (mapped atomicMax)
__global__ void edge_logits(const float* __restrict__ el, const float* __restrict__ er,
                            const int* __restrict__ src, const int* __restrict__ dst,
                            float* __restrict__ eArr, unsigned* __restrict__ mU, int E) {
    int idx = blockIdx.x * blockDim.x + threadIdx.x;     // edge*NH + h
    if (idx >= E * NH) return;
    int e = idx / NH, h = idx - e * NH;
    int s = src[e], d = dst[e];
    float x = el[s * NH + h] + er[d * NH + h];
    x = (x >= 0.f) ? x : 0.2f * x;                       // leaky_relu slope 0.2
    eArr[idx] = x;
    atomicMax(&mU[d * NH + h], mapF(x));
}

// ex = exp(e - m[dst]); segment sum into sF
__global__ void edge_exp(float* __restrict__ eArr, const unsigned* __restrict__ mU,
                         float* __restrict__ sF, const int* __restrict__ dst, int E) {
    int idx = blockIdx.x * blockDim.x + threadIdx.x;
    if (idx >= E * NH) return;
    int e = idx / NH, h = idx - e * NH;
    int d = dst[e];
    float ex = expf(eArr[idx] - unmapF(mU[d * NH + h]));
    eArr[idx] = ex;
    atomicAdd(&sF[d * NH + h], ex);
}

// out[dst] += feat[src] * alpha   (float4-vectorized gather + f32 atomics)
__global__ void edge_scatter(const float* __restrict__ feat, const float* __restrict__ ex,
                             const float* __restrict__ sF, const int* __restrict__ src,
                             const int* __restrict__ dst, float* __restrict__ out, int E) {
    int idx = blockIdx.x * blockDim.x + threadIdx.x;     // edge * (NHD/4)
    if (idx >= E * (NHD / 4)) return;
    int e = idx / (NHD / 4);
    int q = idx - e * (NHD / 4);                         // quad index, f = q*4
    int h = q >> 4;                                      // (q*4)/64
    int s = src[e], d = dst[e];
    float alpha = ex[e * NH + h] / sF[d * NH + h];
    const float4 fv = *(const float4*)(feat + (size_t)s * NHD + q * 4);
    float* o = out + (size_t)d * NHD + q * 4;
    atomicAdd(o + 0, fv.x * alpha);
    atomicAdd(o + 1, fv.y * alpha);
    atomicAdd(o + 2, fv.z * alpha);
    atomicAdd(o + 3, fv.w * alpha);
}

// out = relu(out + bias[f])
__global__ void node_finish(float* __restrict__ out, const float* __restrict__ bias, int n) {
    int i = blockIdx.x * blockDim.x + threadIdx.x;
    if (i >= n) return;
    out[i] = fmaxf(out[i] + bias[i % NHD], 0.f);
}

// ---------------------------------------------------------------------------
// Host-side orchestration
// ---------------------------------------------------------------------------
extern "C" void kernel_launch(void* const* d_in, const int* in_sizes, int n_in,
                              void* d_out, int out_size, void* d_ws, size_t ws_size,
                              hipStream_t stream) {
    (void)in_sizes; (void)n_in; (void)out_size; (void)ws_size;

    const float* features = (const float*)d_in[0];
    const int*   src      = (const int*)d_in[1];
    const int*   dst      = (const int*)d_in[2];
    const float* W1  = (const float*)d_in[3];
    const float* al1 = (const float*)d_in[4];
    const float* ar1 = (const float*)d_in[5];
    const float* b1  = (const float*)d_in[6];
    const float* W2  = (const float*)d_in[7];
    const float* al2 = (const float*)d_in[8];
    const float* ar2 = (const float*)d_in[9];
    const float* b2  = (const float*)d_in[10];
    const float* Wm1 = (const float*)d_in[11];
    const float* bm1 = (const float*)d_in[12];
    const float* Wm2 = (const float*)d_in[13];
    const float* bm2 = (const float*)d_in[14];

    // ---- workspace layout (256B aligned slabs) ----
    char* ws = (char*)d_ws;
    size_t off = 0;
    auto alloc = [&](size_t bytes) -> char* {
        char* p = ws + off;
        off += (bytes + 255) & ~(size_t)255;
        return p;
    };
    __bf16*  Xbf   = (__bf16*)alloc((size_t)NN * FIN * 2);   // 25.6MB, later reused as bfA (19.2MB)
    float*   featA = (float*) alloc((size_t)NN * NHD * 4);   // 38.4MB (feat1/feat2, later x3 overlay)
    float*   featB = (float*) alloc((size_t)NN * NHD * 4);   // 38.4MB (x1/x2 accumulators)
    float*   el    = (float*) alloc((size_t)NN * NH * 4);
    float*   er    = (float*) alloc((size_t)NN * NH * 4);
    unsigned* mU   = (unsigned*)alloc((size_t)NN * NH * 4);
    float*   sF    = (float*) alloc((size_t)NN * NH * 4);
    float*   eArr  = (float*) alloc((size_t)NE * NH * 4);    // 9.6MB
    __bf16*  Wt    = (__bf16*)alloc((size_t)FIN * NHD * 2);  // largest transposed weight
    __bf16*  bfA   = Xbf;                                    // reuse (X dead after GEMM1)
    float*   x3    = featA;                                  // reuse (feat2 dead after layer-2 scatter)
    __bf16*  x3bf  = (__bf16*)((char*)featA + (((size_t)NN * NOUT * 4 + 255) & ~(size_t)255));

    const int TB = 256;
    auto blocks = [](long n, int bs) { return dim3((unsigned)((n + bs - 1) / bs)); };
    auto gemm = [&](const __bf16* A, const __bf16* B, const float* bias, float* C,
                    int M, int N, int K, int act) {
        int nblocks = (M / 16) * (N / 64);                   // one wave per 16x64 block
        gemm_bf16_wmma<<<blocks((long)nblocks * 32, TB), TB, 0, stream>>>(
            A, B, bias, C, M, N, K, act, nblocks);
    };
    auto gat_edges = [&](const float* feat, const float* al, const float* ar,
                         const float* bias, float* outAcc) {
        node_attn  <<<blocks((long)NN * NH, TB), TB, 0, stream>>>(feat, al, ar, el, er, NN);
        fill_zero  <<<blocks((long)NN * NH, TB), TB, 0, stream>>>(mU, NN * NH);
        fill_zero  <<<blocks((long)NN * NH, TB), TB, 0, stream>>>((unsigned*)sF, NN * NH);
        fill_zero  <<<blocks((long)NN * NHD, TB), TB, 0, stream>>>((unsigned*)outAcc, NN * NHD);
        edge_logits<<<blocks((long)NE * NH, TB), TB, 0, stream>>>(el, er, src, dst, eArr, mU, NE);
        edge_exp   <<<blocks((long)NE * NH, TB), TB, 0, stream>>>(eArr, mU, sF, dst, NE);
        edge_scatter<<<blocks((long)NE * (NHD / 4), TB), TB, 0, stream>>>(
            feat, eArr, sF, src, dst, outAcc, NE);
        node_finish<<<blocks((long)NN * NHD, TB), TB, 0, stream>>>(outAcc, bias, NN * NHD);
    };

    // ---- Layer 1: feat1 = X @ W1 ; GAT edge softmax + aggregate -> featB (=x1) ----
    cvt_bf16  <<<blocks((long)NN * FIN, TB), TB, 0, stream>>>(features, Xbf, NN * FIN);
    tconv_bf16<<<blocks((long)FIN * NHD, TB), TB, 0, stream>>>(W1, Wt, FIN, NHD);
    gemm(Xbf, Wt, nullptr, featA, NN, NHD, FIN, 0);
    gat_edges(featA, al1, ar1, b1, featB);

    // ---- Layer 2: feat2 = x1 @ W2 ; GAT -> featB (=x2) ----
    cvt_bf16  <<<blocks((long)NN * NHD, TB), TB, 0, stream>>>(featB, bfA, NN * NHD);
    tconv_bf16<<<blocks((long)NHD * NHD, TB), TB, 0, stream>>>(W2, Wt, NHD, NHD);
    gemm(bfA, Wt, nullptr, featA, NN, NHD, NHD, 0);
    gat_edges(featA, al2, ar2, b2, featB);

    // ---- MLP: x3 = relu(x2 @ Wm1 + bm1) ; out = x3 @ Wm2 + bm2 ----
    cvt_bf16  <<<blocks((long)NN * NHD, TB), TB, 0, stream>>>(featB, bfA, NN * NHD);
    tconv_bf16<<<blocks((long)NHD * NOUT, TB), TB, 0, stream>>>(Wm1, Wt, NHD, NOUT);
    gemm(bfA, Wt, bm1, x3, NN, NOUT, NHD, 1);

    cvt_bf16  <<<blocks((long)NN * NOUT, TB), TB, 0, stream>>>(x3, x3bf, NN * NOUT);
    tconv_bf16<<<blocks((long)NOUT * NOUT, TB), TB, 0, stream>>>(Wm2, Wt, NOUT, NOUT);
    gemm(x3bf, Wt, bm2, (float*)d_out, NN, NOUT, NOUT, 0);
}
